// ScaledLinearDotProductAttention_58952721105439
// MI455X (gfx1250) — matrix operand order repaired and checked by reading, not compiled, not run on hardware
//
#include <hip/hip_runtime.h>

// ---------------- CDNA5 types ----------------
typedef __attribute__((ext_vector_type(16))) __bf16 v16bf;
typedef __attribute__((ext_vector_type(2)))  __bf16 v2bf;
typedef __attribute__((ext_vector_type(8)))  float  v8f;
typedef __attribute__((ext_vector_type(2)))  float  v2f;

struct __align__(16) Half8 { unsigned short h[8]; }; // 16B chunk (8 bf16)
struct __align__(16) U8    { unsigned u[8]; };       // 32B = one v16bf fragment

#define S_LEN 4096
#define DKK   64
#define QTILE 128

// pack two fp32 -> packed bf16 pair. Round-half-up bias + single v_perm_b32.
static __device__ inline unsigned pack2bf(float lo, float hi) {
#if __has_builtin(__builtin_amdgcn_cvt_pk_bf16_f32)
  v2bf v = __builtin_amdgcn_cvt_pk_bf16_f32(lo, hi);
  return __builtin_bit_cast(unsigned, v);
#else
  unsigned ul = __builtin_bit_cast(unsigned, lo) + 0x8000u;
  unsigned uh = __builtin_bit_cast(unsigned, hi) + 0x8000u;
  return __builtin_amdgcn_perm(uh, ul, 0x07060302u);  // {uh[31:16], ul[31:16]}
#endif
}
// truncating pack (for P in [0,1]: precision non-critical, 1 VALU op)
static __device__ inline unsigned pack2bf_t(float lo, float hi) {
  return __builtin_amdgcn_perm(__builtin_bit_cast(unsigned, hi),
                               __builtin_bit_cast(unsigned, lo), 0x07060302u);
}
static __device__ inline float fast_exp2(float x) {
#if __has_builtin(__builtin_amdgcn_exp2f)
  return __builtin_amdgcn_exp2f(x);
#else
  return exp2f(x);
#endif
}

// ds_swizzle SWAPX16 (xor 0x10 within wave32): offset = (0x10<<10) | 0x1f
#define SWZF(v) __builtin_bit_cast(float, __builtin_amdgcn_ds_swizzle(__builtin_bit_cast(int, (v)), 0x401f))

__global__ __launch_bounds__(256) void fa_causal_bf16wmma_t4(
    const float* __restrict__ ql, const float* __restrict__ kl,
    const float* __restrict__ vl, float* __restrict__ out) {
  __shared__ unsigned short sK[2][64 * 64];   // [buf][key][d]  bf16, 2x8KB
  __shared__ unsigned short sV[2][64 * 64];   // [buf][d][key]  bf16 (transposed), 2x8KB

  const int tid  = threadIdx.x;
  const int wave = tid >> 5;
  const int lane = tid & 31;
  const int hf   = lane >> 4;   // lane half (0/1)
  const int nn   = lane & 15;   // query column
  // permuted key row so P relayout is in-lane:
  // tile t2 row M=nn -> key = (t2>>1)*32 + ((nn&8)<<1 | (nn&7)) + (t2&1)*8
  const int keysel = ((nn & 8) << 1) | (nn & 7);

  const int nqblk = S_LEN / QTILE;
  const int b   = blockIdx.x / nqblk;
  const int q0  = (blockIdx.x % nqblk) * QTILE;
  const int wq0 = q0 + wave * 16;
  const int qrow = wq0 + nn;

  // fold 1/sqrt(dk) * log2(e) into Q so softmax is pure exp2 (v_exp_f32)
  const float SCALE = 0.125f * 1.44269504088896340736f;

  // ---------------- Q -> two B-fragments (Q^T: K=d, N=query) ----------------
  v16bf bq[2];
  {
    const float* qbase = ql + ((size_t)b * S_LEN + qrow) * DKK;
#pragma unroll
    for (int c = 0; c < 2; ++c) {
      const float* p = qbase + c * 32 + hf * 8;
      U8 fr;
#pragma unroll
      for (int i = 0; i < 4; ++i) {
        const float4 x = *(const float4*)(p + (i >> 1) * 16 + (i & 1) * 4);
        fr.u[i * 2 + 0] = pack2bf(x.x * SCALE, x.y * SCALE);
        fr.u[i * 2 + 1] = pack2bf(x.z * SCALE, x.w * SCALE);
      }
      bq[c] = __builtin_bit_cast(v16bf, fr);
    }
  }

  // ---------------- staging geometry ----------------
  const int skey = tid >> 2;            // K: one key, 16 d
  const int sd0  = (tid & 3) * 16;
  const int kp2  = tid & 31;            // V: key pair 2*kp2, 2*kp2+1, 8 d
  const int vd0  = (tid >> 5) * 8;
  const float* kg = kl + (size_t)b * S_LEN * DKK;
  const float* vg = vl + (size_t)b * S_LEN * DKK;

  float4 kreg[4], vreg[4];
  auto issue_loads = [&](int kb) {
    const float* kp = kg + (size_t)(kb + skey) * DKK + sd0;
#pragma unroll
    for (int c = 0; c < 4; ++c) kreg[c] = *(const float4*)(kp + c * 4);
    const float* vp = vg + (size_t)(kb + 2 * kp2) * DKK + vd0;
    vreg[0] = *(const float4*)(vp);
    vreg[1] = *(const float4*)(vp + 4);
    vreg[2] = *(const float4*)(vp + DKK);
    vreg[3] = *(const float4*)(vp + DKK + 4);
  };
  auto store_lds = [&](int buf) {
    unsigned short* K = &sK[buf][0];
#pragma unroll
    for (int c = 0; c < 4; ++c) {
      *(unsigned*)&K[skey * 64 + sd0 + c * 4 + 0] = pack2bf(kreg[c].x, kreg[c].y);
      *(unsigned*)&K[skey * 64 + sd0 + c * 4 + 2] = pack2bf(kreg[c].z, kreg[c].w);
    }
    unsigned short* V = &sV[buf][0];   // transposed: [d][key], pack key pair per dword
    *(unsigned*)&V[(vd0 + 0) * 64 + 2 * kp2] = pack2bf(vreg[0].x, vreg[2].x);
    *(unsigned*)&V[(vd0 + 1) * 64 + 2 * kp2] = pack2bf(vreg[0].y, vreg[2].y);
    *(unsigned*)&V[(vd0 + 2) * 64 + 2 * kp2] = pack2bf(vreg[0].z, vreg[2].z);
    *(unsigned*)&V[(vd0 + 3) * 64 + 2 * kp2] = pack2bf(vreg[0].w, vreg[2].w);
    *(unsigned*)&V[(vd0 + 4) * 64 + 2 * kp2] = pack2bf(vreg[1].x, vreg[3].x);
    *(unsigned*)&V[(vd0 + 5) * 64 + 2 * kp2] = pack2bf(vreg[1].y, vreg[3].y);
    *(unsigned*)&V[(vd0 + 6) * 64 + 2 * kp2] = pack2bf(vreg[1].z, vreg[3].z);
    *(unsigned*)&V[(vd0 + 7) * 64 + 2 * kp2] = pack2bf(vreg[1].w, vreg[3].w);
  };

  // ---------------- flash-attention state (per-lane scalars) ----------------
  v8f acc[4];                                // O^T: d = t*16 + hf*8 + r, q = nn
#pragma unroll
  for (int t = 0; t < 4; ++t) acc[t] = (v8f){};
  float m = -3.0e38f, l = 0.0f;

  const int kend = q0 + QTILE;
  issue_loads(0);
  int buf = 0;
  for (int kb = 0; kb < kend; kb += 64) {
    store_lds(buf);
    __syncthreads();                          // staging of buf visible to all waves
    if (kb + 64 < kend) issue_loads(kb + 64); // prefetch next tile under compute

    const unsigned short* K = &sK[buf][0];
    const unsigned short* V = &sV[buf][0];
    buf ^= 1;

    if (kb > wq0 + 15) continue;              // whole tile above diagonal (uniform per wave)
    const bool doHi = (kb + 32) <= (wq0 + 15);

    // ---- S^T = K * Q^T : four permuted 16x16 tiles over 64 keys ------------
    v8f sc[4];
#pragma unroll
    for (int t2 = 0; t2 < 2; ++t2) {
      sc[t2] = (v8f){};
      const int keyrow = keysel + t2 * 8;
#pragma unroll
      for (int c = 0; c < 2; ++c) {
        U8 a;
        *(Half8*)&a.u[0] = *(const Half8*)&K[keyrow * 64 + c * 32 + hf * 8];
        *(Half8*)&a.u[4] = *(const Half8*)&K[keyrow * 64 + c * 32 + 16 + hf * 8];
        sc[t2] = __builtin_amdgcn_wmma_f32_16x16x32_bf16(
            false, __builtin_bit_cast(v16bf, a), false, bq[c], (short)0, sc[t2], false, false);
      }
    }
    if (doHi) {
#pragma unroll
      for (int t2 = 2; t2 < 4; ++t2) {
        sc[t2] = (v8f){};
        const int keyrow = 32 + keysel + (t2 & 1) * 8;
#pragma unroll
        for (int c = 0; c < 2; ++c) {
          U8 a;
          *(Half8*)&a.u[0] = *(const Half8*)&K[keyrow * 64 + c * 32 + hf * 8];
          *(Half8*)&a.u[4] = *(const Half8*)&K[keyrow * 64 + c * 32 + 16 + hf * 8];
          sc[t2] = __builtin_amdgcn_wmma_f32_16x16x32_bf16(
              false, __builtin_bit_cast(v16bf, a), false, bq[c], (short)0, sc[t2], false, false);
        }
      }
    }

    // C layout: tile0[r]=key kb+hf*16+r, tile1=+8, tile2=+32, tile3=+40
    float v0[8], v1[8], v2[8], v3[8];
#pragma unroll
    for (int r = 0; r < 8; ++r) { v0[r] = sc[0][r]; v1[r] = sc[1][r]; }
    if (doHi) {
#pragma unroll
      for (int r = 0; r < 8; ++r) { v2[r] = sc[2][r]; v3[r] = sc[3][r]; }
    } else {
#pragma unroll
      for (int r = 0; r < 8; ++r) { v2[r] = -3.0e38f; v3[r] = -3.0e38f; }
    }
    if (kb + 63 > wq0) {                      // diagonal tile: causal mask
      const int lim = qrow - kb - hf * 16;    // valid iff key offset <= lim
#pragma unroll
      for (int r = 0; r < 8; ++r) {
        if (r > lim)      v0[r] = -3.0e38f;
        if (r + 8 > lim)  v1[r] = -3.0e38f;
        if (r + 32 > lim) v2[r] = -3.0e38f;
        if (r + 40 > lim) v3[r] = -3.0e38f;
      }
    }

    // ---- online softmax over 64 keys (32 in-lane + xor16 half) -------------
    float mA[8];
#pragma unroll
    for (int r = 0; r < 8; ++r) mA[r] = fmaxf(fmaxf(v0[r], v1[r]), fmaxf(v2[r], v3[r]));
    float mx = fmaxf(fmaxf(fmaxf(mA[0], mA[1]), fmaxf(mA[2], mA[3])),
                     fmaxf(fmaxf(mA[4], mA[5]), fmaxf(mA[6], mA[7])));
    mx = fmaxf(mx, SWZF(mx));
    const float mnew = fmaxf(m, mx);
    const float alpha = fast_exp2(m - mnew);
    m = mnew;

    float p0[8], p1[8], p2[8], p3[8];
#pragma unroll
    for (int r = 0; r < 8; ++r) {
      p0[r] = fast_exp2(v0[r] - mnew);
      p1[r] = fast_exp2(v1[r] - mnew);
      p2[r] = fast_exp2(v2[r] - mnew);
      p3[r] = fast_exp2(v3[r] - mnew);
    }
    // packed-f32 sum tree (v_pk_add_f32)
    v2f q01[4], q23[4];
#pragma unroll
    for (int i = 0; i < 4; ++i) {
      q01[i] = (v2f){p0[2 * i], p0[2 * i + 1]} + (v2f){p1[2 * i], p1[2 * i + 1]};
      q23[i] = (v2f){p2[2 * i], p2[2 * i + 1]} + (v2f){p3[2 * i], p3[2 * i + 1]};
    }
    const v2f qs = ((q01[0] + q01[1]) + (q01[2] + q01[3])) +
                   ((q23[0] + q23[1]) + (q23[2] + q23[3]));
    float ssum = qs[0] + qs[1];
    ssum += SWZF(ssum);
    l = l * alpha + ssum;

    // ---- rescale running O^T (packed-f32 muls, once per 64 keys) ----
    const v8f av = {alpha, alpha, alpha, alpha, alpha, alpha, alpha, alpha};
#pragma unroll
    for (int t = 0; t < 4; ++t) acc[t] *= av;

    // ---- P B-fragments: pure in-lane truncating packs (1 v_perm each) ------
    U8 w0, w1;
#pragma unroll
    for (int i = 0; i < 4; ++i) {
      w0.u[i]     = pack2bf_t(p0[2 * i], p0[2 * i + 1]);  // keys hf*16 + 0..7
      w0.u[4 + i] = pack2bf_t(p1[2 * i], p1[2 * i + 1]);  // keys hf*16 + 8..15
      w1.u[i]     = pack2bf_t(p2[2 * i], p2[2 * i + 1]);  // keys 32 + hf*16 + 0..7
      w1.u[4 + i] = pack2bf_t(p3[2 * i], p3[2 * i + 1]);  // keys 32 + hf*16 + 8..15
    }
    const v16bf pb0 = __builtin_bit_cast(v16bf, w0);
    const v16bf pb1 = __builtin_bit_cast(v16bf, w1);

    // ---- O^T += V^T * P : keys 0-31 always, 32-63 only below diagonal ------
#pragma unroll
    for (int t = 0; t < 4; ++t) {
      U8 a;
      *(Half8*)&a.u[0] = *(const Half8*)&V[(t * 16 + nn) * 64 + hf * 16];
      *(Half8*)&a.u[4] = *(const Half8*)&V[(t * 16 + nn) * 64 + 8 + hf * 16];
      acc[t] = __builtin_amdgcn_wmma_f32_16x16x32_bf16(
          false, __builtin_bit_cast(v16bf, a), false, pb0, (short)0, acc[t], false, false);
    }
    if (doHi) {
#pragma unroll
      for (int t = 0; t < 4; ++t) {
        U8 a;
        *(Half8*)&a.u[0] = *(const Half8*)&V[(t * 16 + nn) * 64 + 32 + hf * 16];
        *(Half8*)&a.u[4] = *(const Half8*)&V[(t * 16 + nn) * 64 + 40 + hf * 16];
        acc[t] = __builtin_amdgcn_wmma_f32_16x16x32_bf16(
            false, __builtin_bit_cast(v16bf, a), false, pb1, (short)0, acc[t], false, false);
      }
    }
  }

  // ---------------- epilogue: vector 1/l scale, 8 contiguous d per tile ----
  const float inv = 1.0f / l;
  const v8f iv = {inv, inv, inv, inv, inv, inv, inv, inv};
  float* obase = out + ((size_t)b * S_LEN + qrow) * DKK;
#pragma unroll
  for (int t = 0; t < 4; ++t) {
    const v8f o = acc[t] * iv;               // v_pk_mul_f32
    float4 o0 = make_float4(o[0], o[1], o[2], o[3]);
    float4 o1 = make_float4(o[4], o[5], o[6], o[7]);
    *(float4*)(obase + t * 16 + hf * 8 + 0) = o0;
    *(float4*)(obase + t * 16 + hf * 8 + 4) = o1;
  }
}

extern "C" void kernel_launch(void* const* d_in, const int* in_sizes, int n_in,
                              void* d_out, int out_size, void* d_ws, size_t ws_size,
                              hipStream_t stream) {
  const float* ql = (const float*)d_in[0];
  const float* kl = (const float*)d_in[1];
  const float* vl = (const float*)d_in[2];
  float* out = (float*)d_out;

  const int B = in_sizes[0] / (S_LEN * DKK);       // 16
  dim3 grid(B * (S_LEN / QTILE));                  // 512 blocks
  dim3 block(256);                                 // 8 waves (wave32)
  fa_causal_bf16wmma_t4<<<grid, block, 0, stream>>>(ql, kl, vl, out);
}
